// LAM_29832842838412
// MI455X (gfx1250) — matrix operand order repaired and checked by reading, not compiled, not run on hardware
//
#include <hip/hip_runtime.h>
#include <hip/hip_bf16.h>
#include <cstdint>

// ---------------------------------------------------------------------------
// Types for CDNA5 WMMA
// ---------------------------------------------------------------------------
typedef __attribute__((ext_vector_type(16))) __bf16 v16bf;
typedef __attribute__((ext_vector_type(8)))  float  v8f;
typedef __attribute__((ext_vector_type(4)))  int    v4i;

#define GEMM_RELU    1
#define GEMM_ADDPOS  2
#define GEMM_OUTBF16 4

// ---------------------------------------------------------------------------
// Async global->LDS copy (CDNA5 GLOBAL_LOAD_ASYNC_TO_LDS_B128, ASYNCcnt),
// guarded so we degrade to a manual copy if the builtin is absent.
// Builtin signature (from hipcc diagnostic): arg0 = v4i* in device(global) AS,
// arg1 = v4i* in shared(LDS) AS, arg2 = imm offset, arg3 = cpol.
// ---------------------------------------------------------------------------
#define AS1 __attribute__((address_space(1)))
#define AS3 __attribute__((address_space(3)))

#if defined(__has_builtin)
#if __has_builtin(__builtin_amdgcn_global_load_async_to_lds_b128)
#define ASYNC_OK 1
#endif
#endif
#ifndef ASYNC_OK
#define ASYNC_OK 0
#endif

__device__ __forceinline__ void cp_async16(const void* g, void* l)
{
#if ASYNC_OK
    __builtin_amdgcn_global_load_async_to_lds_b128(
        (AS1 v4i*)(uintptr_t)g, (AS3 v4i*)(uintptr_t)l, 0, 0);
#else
    *(int4*)l = *(const int4*)g;
#endif
}

__device__ __forceinline__ void wait_async_lds()
{
#if ASYNC_OK
#if defined(__has_builtin) && __has_builtin(__builtin_amdgcn_s_wait_asynccnt)
    __builtin_amdgcn_s_wait_asynccnt(0);
#else
    asm volatile("s_wait_asynccnt 0x0" ::: "memory");
#endif
#endif
}

// ---------------------------------------------------------------------------
// bf16 WMMA GEMM:  C[M,N] = A[M,K](bf16) @ B[K,N](fp32->bf16) + bias
// Block = 256 threads (8 wave32s). Tile = 128x128, BK=32, double-buffered LDS.
// Wave (wr 0..3, wc 0..1): 32 rows x 64 cols = 2 A-frags x 4 B-frags
// -> 8 v_wmma_f32_16x16x32_bf16 per K-step.
// ---------------------------------------------------------------------------
__global__ __launch_bounds__(256)
void gemm_bf16_wmma(const __bf16* __restrict__ A, const float* __restrict__ B,
                    const float* __restrict__ bias, float* __restrict__ Cf,
                    __bf16* __restrict__ Cbf, const float* __restrict__ pos,
                    int M, int N, int Kdim, int flags)
{
    __shared__ __bf16 sA[2][128 * 32];   // [row][k], straight copy of A slab
    __shared__ __bf16 sB[2][128 * 32];   // transposed: [n][k]

    const int tid  = threadIdx.x;
    const int wid  = tid >> 5;
    const int lane = tid & 31;
    const int mBase = blockIdx.y * 128;
    const int nBase = blockIdx.x * 128;
    const int wr   = wid >> 1;       // 0..3 : 32-row group
    const int wc   = wid & 1;        // 0..1 : 64-col group
    const int half = lane >> 4;      // 0/1
    const int l15  = lane & 15;

    v8f acc[2][4];
#pragma unroll
    for (int mt = 0; mt < 2; ++mt)
#pragma unroll
        for (int nt = 0; nt < 4; ++nt)
            acc[mt][nt] = (v8f){0.f,0.f,0.f,0.f,0.f,0.f,0.f,0.f};

    auto stage = [&](int buf, int k0) {
        // A slab: 128 rows x 32 k of bf16 = 8KB = 512 x 16B chunks (async DMA)
        for (int c = tid; c < 512; c += 256) {
            int row = c >> 2, seg = c & 3;
            const char* g = (const char*)(A + (size_t)(mBase + row) * Kdim + k0)
                            + seg * 16;
            cp_async16(g, (char*)&sA[buf][0] + c * 16);
        }
        // B slab: 32 k x 128 n fp32 -> bf16, stored transposed [n][k]
        for (int i = tid; i < 4096; i += 256) {
            int k = i >> 7, n = i & 127;
            sB[buf][n * 32 + k] = (__bf16)B[(size_t)(k0 + k) * N + (nBase + n)];
        }
    };

    stage(0, 0);
    wait_async_lds();
    __syncthreads();

    int buf = 0;
    for (int k0 = 0; k0 < Kdim; k0 += 32) {
        if (k0 + 32 < Kdim) stage(buf ^ 1, k0 + 32);   // prefetch next slice

        union F { v16bf v; unsigned u[8]; } a[2], b[4];
        const __bf16* sAb = &sA[buf][0];
        const __bf16* sBb = &sB[buf][0];
        // A fragment: lanes 0-15 K=0..15, lanes 16-31 K=16..31 (contiguous)
#pragma unroll
        for (int mt = 0; mt < 2; ++mt) {
            const unsigned* arow =
                (const unsigned*)(sAb + (wr * 32 + mt * 16 + l15) * 32);
#pragma unroll
            for (int j = 0; j < 8; ++j)
                a[mt].u[j] = arow[half * 8 + j];
        }
        // B fragment: lanes 0-15 K=0..15, lanes 16-31 K=16..31 (contiguous)
#pragma unroll
        for (int nt = 0; nt < 4; ++nt) {
            const unsigned* bcol =
                (const unsigned*)(sBb + (wc * 64 + nt * 16 + l15) * 32);
#pragma unroll
            for (int j = 0; j < 8; ++j)
                b[nt].u[j] = bcol[half * 8 + j];
        }
#pragma unroll
        for (int mt = 0; mt < 2; ++mt)
#pragma unroll
            for (int nt = 0; nt < 4; ++nt)
                acc[mt][nt] = __builtin_amdgcn_wmma_f32_16x16x32_bf16(
                    false, a[mt].v, false, b[nt].v, (short)0,
                    acc[mt][nt], false, false);

        wait_async_lds();
        __syncthreads();
        buf ^= 1;
    }

    // C layout: VGPR i -> M = half*8 + i ; N = l15 (+ tile bases)
#pragma unroll
    for (int mt = 0; mt < 2; ++mt) {
#pragma unroll
        for (int nt = 0; nt < 4; ++nt) {
            int n  = nBase + wc * 64 + nt * 16 + l15;
            int mb = mBase + wr * 32 + mt * 16 + half * 8;
            float bn = bias[n];
#pragma unroll
            for (int i = 0; i < 8; ++i) {
                int m = mb + i;
                float v = acc[mt][nt][i] + bn;
                if (flags & GEMM_ADDPOS) v += pos[(m & 15) * 512 + n];
                if (flags & GEMM_RELU)   v = fmaxf(v, 0.f);
                if (flags & GEMM_OUTBF16) Cbf[(size_t)m * N + n] = (__bf16)v;
                else                      Cf[(size_t)m * N + n] = v;
            }
        }
    }
}

// ---------------------------------------------------------------------------
// Patch extraction: frames [16,64,3,64,64] -> bf16 patches [16384, 768]
// ---------------------------------------------------------------------------
__global__ __launch_bounds__(256)
void patchify(const float* __restrict__ frames, __bf16* __restrict__ out)
{
    int m = blockIdx.x;
    int b = m >> 10;
    int rem = m & 1023;
    int s = rem >> 4, p = rem & 15;
    int ph = p >> 2, pw = p & 3;
    for (int idx = threadIdx.x; idx < 768; idx += 256) {
        int c = idx >> 8, r = idx & 255;
        int p1 = r >> 4, p2 = r & 15;
        size_t fo = ((((size_t)b * 64 + s) * 3 + c) * 64 + (ph * 16 + p1)) * 64
                    + (pw * 16 + p2);
        out[(size_t)m * 768 + idx] = (__bf16)frames[fo];
    }
}

__global__ __launch_bounds__(256)
void f32_to_bf16(const float* __restrict__ in, __bf16* __restrict__ out, int n)
{
    int i = blockIdx.x * 256 + threadIdx.x;
    if (i < n) out[i] = (__bf16)in[i];
}

__global__ __launch_bounds__(256)
void copy_f32(const float* __restrict__ in, float* __restrict__ out, int n)
{
    int i = blockIdx.x * 256 + threadIdx.x;
    if (i < n) out[i] = in[i];
}

// ---------------------------------------------------------------------------
// Spatial attention: block per (b,s) frame. 16 tokens, 8 heads, d=64.
// ---------------------------------------------------------------------------
__global__ __launch_bounds__(256)
void attn_spatial(const float* __restrict__ Q, const float* __restrict__ K,
                  const float* __restrict__ V, float* __restrict__ O)
{
    __shared__ float sS[256];
    int bs = blockIdx.x;
    int tid = threadIdx.x;
    const float scale = 0.125f;  // 1/sqrt(64)
    int i = tid >> 4, j = tid & 15;
    for (int h = 0; h < 8; ++h) {
        size_t qo = ((size_t)bs * 16 + i) * 512 + h * 64;
        size_t ko = ((size_t)bs * 16 + j) * 512 + h * 64;
        float acc = 0.f;
        for (int d = 0; d < 64; ++d) acc += Q[qo + d] * K[ko + d];
        sS[i * 16 + j] = acc * scale;
        __syncthreads();
        if (tid < 16) {
            float mx = -1e30f;
            for (int jj = 0; jj < 16; ++jj) mx = fmaxf(mx, sS[tid * 16 + jj]);
            float sum = 0.f;
            for (int jj = 0; jj < 16; ++jj) {
                float e = __expf(sS[tid * 16 + jj] - mx);
                sS[tid * 16 + jj] = e; sum += e;
            }
            float inv = 1.f / sum;
            for (int jj = 0; jj < 16; ++jj) sS[tid * 16 + jj] *= inv;
        }
        __syncthreads();
        int dc = (tid & 15) * 4;
        float o0 = 0.f, o1 = 0.f, o2 = 0.f, o3 = 0.f;
        for (int jj = 0; jj < 16; ++jj) {
            float w = sS[i * 16 + jj];
            size_t vo = ((size_t)bs * 16 + jj) * 512 + h * 64 + dc;
            o0 += w * V[vo]; o1 += w * V[vo + 1];
            o2 += w * V[vo + 2]; o3 += w * V[vo + 3];
        }
        size_t oo = ((size_t)bs * 16 + i) * 512 + h * 64 + dc;
        O[oo] = o0; O[oo + 1] = o1; O[oo + 2] = o2; O[oo + 3] = o3;
        __syncthreads();
    }
}

// ---------------------------------------------------------------------------
// Temporal causal attention: block per (b, p, h). 64 timesteps, d=64.
// ---------------------------------------------------------------------------
__global__ __launch_bounds__(256)
void attn_temporal(const float* __restrict__ Q, const float* __restrict__ K,
                   const float* __restrict__ V, float* __restrict__ O)
{
    __shared__ float sS[64 * 64];
    int bid = blockIdx.x;
    int h = bid & 7, p = (bid >> 3) & 15, b = bid >> 7;
    int tid = threadIdx.x;
    const float scale = 0.125f;
    int i = tid >> 2;                 // query time
    int jc = (tid & 3) * 16;
    size_t qo = (((size_t)b * 64 + i) * 16 + p) * 512 + h * 64;
    for (int jj = 0; jj < 16; ++jj) {
        int j = jc + jj;
        float acc = 0.f;
        size_t ko = (((size_t)b * 64 + j) * 16 + p) * 512 + h * 64;
        for (int d = 0; d < 64; ++d) acc += Q[qo + d] * K[ko + d];
        sS[i * 64 + j] = (j <= i) ? acc * scale : -1e30f;
    }
    __syncthreads();
    if (tid < 64) {
        float mx = -1e30f;
        for (int j = 0; j <= tid; ++j) mx = fmaxf(mx, sS[tid * 64 + j]);
        float sum = 0.f;
        for (int j = 0; j < 64; ++j) {
            float e = __expf(sS[tid * 64 + j] - mx);
            sS[tid * 64 + j] = e; sum += e;
        }
        float inv = 1.f / sum;
        for (int j = 0; j < 64; ++j) sS[tid * 64 + j] *= inv;
    }
    __syncthreads();
    int dc = (tid & 3) * 16;
    float o[16];
#pragma unroll
    for (int d = 0; d < 16; ++d) o[d] = 0.f;
    for (int j = 0; j <= i; ++j) {
        float w = sS[i * 64 + j];
        size_t vo = (((size_t)b * 64 + j) * 16 + p) * 512 + h * 64 + dc;
#pragma unroll
        for (int d = 0; d < 16; ++d) o[d] += w * V[vo + d];
    }
    size_t oo = (((size_t)b * 64 + i) * 16 + p) * 512 + h * 64 + dc;
#pragma unroll
    for (int d = 0; d < 16; ++d) O[oo + d] = o[d];
}

// ---------------------------------------------------------------------------
// Residual + LayerNorm over E=512: xout = LN(xin + delta) * g + b
// ---------------------------------------------------------------------------
__global__ __launch_bounds__(256)
void ln_residual(const float* __restrict__ xin, const float* __restrict__ delta,
                 float* __restrict__ xout, const float* __restrict__ g,
                 const float* __restrict__ bv)
{
    __shared__ float sv[512];
    __shared__ float red[256];
    int m = blockIdx.x, tid = threadIdx.x;
    float local = 0.f;
    for (int e = tid; e < 512; e += 256) {
        float v = xin[(size_t)m * 512 + e] + delta[(size_t)m * 512 + e];
        sv[e] = v; local += v;
    }
    red[tid] = local; __syncthreads();
    for (int s = 128; s > 0; s >>= 1) {
        if (tid < s) red[tid] += red[tid + s];
        __syncthreads();
    }
    float mean = red[0] * (1.f / 512.f);
    __syncthreads();
    local = 0.f;
    for (int e = tid; e < 512; e += 256) { float d = sv[e] - mean; local += d * d; }
    red[tid] = local; __syncthreads();
    for (int s = 128; s > 0; s >>= 1) {
        if (tid < s) red[tid] += red[tid + s];
        __syncthreads();
    }
    float rstd = rsqrtf(red[0] * (1.f / 512.f) + 1e-5f);
    for (int e = tid; e < 512; e += 256)
        xout[(size_t)m * 512 + e] = (sv[e] - mean) * rstd * g[e] + bv[e];
}

// ---------------------------------------------------------------------------
// Mean pool over the 16 patches: X [1024,16,512] -> pooled [1024,512]
// ---------------------------------------------------------------------------
__global__ __launch_bounds__(256)
void pool_mean(const float* __restrict__ X, float* __restrict__ pooled)
{
    int bs = blockIdx.x;
    for (int e = threadIdx.x; e < 512; e += 256) {
        float s = 0.f;
        for (int p = 0; p < 16; ++p)
            s += X[((size_t)bs * 16 + p) * 512 + e];
        pooled[(size_t)bs * 512 + e] = s * (1.f / 16.f);
    }
}

// ---------------------------------------------------------------------------
// Head: pairs = [pooled[b,t], pooled[b,t+1]] (1024) -> LN -> @ h_w + h_b
// ---------------------------------------------------------------------------
__global__ __launch_bounds__(256)
void head_kernel(const float* __restrict__ pooled, const float* __restrict__ g,
                 const float* __restrict__ bv, const float* __restrict__ hw,
                 const float* __restrict__ hb, float* __restrict__ actions)
{
    __shared__ float sp[1024];
    __shared__ float red[256];
    int b = blockIdx.x / 63, t = blockIdx.x % 63;
    int tid = threadIdx.x;
    float local = 0.f;
    for (int i = tid; i < 1024; i += 256) {
        float v = (i < 512)
            ? pooled[((size_t)b * 64 + t) * 512 + i]
            : pooled[((size_t)b * 64 + t + 1) * 512 + (i - 512)];
        sp[i] = v; local += v;
    }
    red[tid] = local; __syncthreads();
    for (int s = 128; s > 0; s >>= 1) {
        if (tid < s) red[tid] += red[tid + s];
        __syncthreads();
    }
    float mean = red[0] * (1.f / 1024.f);
    __syncthreads();
    local = 0.f;
    for (int i = tid; i < 1024; i += 256) { float d = sp[i] - mean; local += d * d; }
    red[tid] = local; __syncthreads();
    for (int s = 128; s > 0; s >>= 1) {
        if (tid < s) red[tid] += red[tid + s];
        __syncthreads();
    }
    float rstd = rsqrtf(red[0] * (1.f / 1024.f) + 1e-5f);
    __syncthreads();
    for (int i = tid; i < 1024; i += 256)
        sp[i] = (sp[i] - mean) * rstd * g[i] + bv[i];
    __syncthreads();
    if (tid < 64) {
        float acc = hb[tid];
        for (int e = 0; e < 1024; ++e) acc += sp[e] * hw[e * 64 + tid];
        actions[((size_t)b * 63 + t) * 64 + tid] = acc;
    }
}

// ---------------------------------------------------------------------------
// Host launcher
// ---------------------------------------------------------------------------
extern "C" void kernel_launch(void* const* d_in, const int* in_sizes, int n_in,
                              void* d_out, int out_size, void* d_ws, size_t ws_size,
                              hipStream_t stream)
{
    const int E = 512, FF = 2048, L = 6;
    const int M = 16384;                 // 16*64*16 tokens
    const float* frames  = (const float*)d_in[0];
    const float* w_patch = (const float*)d_in[1];
    const float* b_patch = (const float*)d_in[2];
    const float* pos     = (const float*)d_in[3];
    const float* s_wq = (const float*)d_in[4];
    const float* s_wk = (const float*)d_in[5];
    const float* s_wv = (const float*)d_in[6];
    const float* s_wo = (const float*)d_in[7];
    const float* s_bq = (const float*)d_in[8];
    const float* s_bk = (const float*)d_in[9];
    const float* s_bv = (const float*)d_in[10];
    const float* s_bo = (const float*)d_in[11];
    const float* s_lng = (const float*)d_in[12];
    const float* s_lnb = (const float*)d_in[13];
    const float* t_wq = (const float*)d_in[14];
    const float* t_wk = (const float*)d_in[15];
    const float* t_wv = (const float*)d_in[16];
    const float* t_wo = (const float*)d_in[17];
    const float* t_bq = (const float*)d_in[18];
    const float* t_bk = (const float*)d_in[19];
    const float* t_bv = (const float*)d_in[20];
    const float* t_bo = (const float*)d_in[21];
    const float* t_lng = (const float*)d_in[22];
    const float* t_lnb = (const float*)d_in[23];
    const float* f_w1 = (const float*)d_in[24];
    const float* f_b1 = (const float*)d_in[25];
    const float* f_w2 = (const float*)d_in[26];
    const float* f_b2 = (const float*)d_in[27];
    const float* f_lng = (const float*)d_in[28];
    const float* f_lnb = (const float*)d_in[29];
    const float* h_lng = (const float*)d_in[30];
    const float* h_lnb = (const float*)d_in[31];
    const float* h_w   = (const float*)d_in[32];
    const float* h_b   = (const float*)d_in[33];

    // Workspace carve (sizes all multiples of 256B, no padding needed)
    char* base = (char*)d_ws;
    const size_t XB = (size_t)M * E * sizeof(float);          // 33,554,432 B
    float* X  = (float*)base;            base += XB;
    float* Qf = (float*)base;            base += XB;
    float* Kf = (float*)base;            base += XB;   // Qf..Kf contiguous
    float* Vf = (float*)base;            base += XB;
    float* Of = (float*)base;            base += XB;
    float* pooled = (float*)base;        base += (size_t)1024 * E * sizeof(float);
    __bf16* T0 = (__bf16*)base;          base += (size_t)M * FF * sizeof(__bf16);
    __bf16* H1bf = (__bf16*)Qf;          // 67 MB spanning Q+K (dead during FFN)

    dim3 blk(256);
    const int nElem = M * E;                  // 8,388,608
    const int cvtGrid = (nElem + 255) / 256;  // 32768
    const dim3 gE(E / 128, M / 128);          // GEMMs with N=512
    const dim3 gF(FF / 128, M / 128);         // GEMM with N=2048

    // ---- patch embed: X = patches @ w_patch + b_patch + pos ----
    patchify<<<M, blk, 0, stream>>>(frames, T0);
    gemm_bf16_wmma<<<gE, blk, 0, stream>>>(
        T0, w_patch, b_patch, X, nullptr, pos, M, E, 768, GEMM_ADDPOS);

    for (int l = 0; l < L; ++l) {
        // ===== spatial attention =====
        f32_to_bf16<<<cvtGrid, blk, 0, stream>>>(X, T0, nElem);
        gemm_bf16_wmma<<<gE, blk, 0, stream>>>(
            T0, s_wq + (size_t)l * E * E, s_bq + (size_t)l * E,
            Qf, nullptr, nullptr, M, E, E, 0);
        gemm_bf16_wmma<<<gE, blk, 0, stream>>>(
            T0, s_wk + (size_t)l * E * E, s_bk + (size_t)l * E,
            Kf, nullptr, nullptr, M, E, E, 0);
        gemm_bf16_wmma<<<gE, blk, 0, stream>>>(
            T0, s_wv + (size_t)l * E * E, s_bv + (size_t)l * E,
            Vf, nullptr, nullptr, M, E, E, 0);
        attn_spatial<<<1024, blk, 0, stream>>>(Qf, Kf, Vf, Of);
        f32_to_bf16<<<cvtGrid, blk, 0, stream>>>(Of, T0, nElem);
        gemm_bf16_wmma<<<gE, blk, 0, stream>>>(
            T0, s_wo + (size_t)l * E * E, s_bo + (size_t)l * E,
            Qf, nullptr, nullptr, M, E, E, 0);     // P -> Qf
        ln_residual<<<M, blk, 0, stream>>>(X, Qf, X,
            s_lng + (size_t)l * E, s_lnb + (size_t)l * E);

        // ===== temporal attention (causal) =====
        f32_to_bf16<<<cvtGrid, blk, 0, stream>>>(X, T0, nElem);
        gemm_bf16_wmma<<<gE, blk, 0, stream>>>(
            T0, t_wq + (size_t)l * E * E, t_bq + (size_t)l * E,
            Qf, nullptr, nullptr, M, E, E, 0);
        gemm_bf16_wmma<<<gE, blk, 0, stream>>>(
            T0, t_wk + (size_t)l * E * E, t_bk + (size_t)l * E,
            Kf, nullptr, nullptr, M, E, E, 0);
        gemm_bf16_wmma<<<gE, blk, 0, stream>>>(
            T0, t_wv + (size_t)l * E * E, t_bv + (size_t)l * E,
            Vf, nullptr, nullptr, M, E, E, 0);
        attn_temporal<<<16 * 16 * 8, blk, 0, stream>>>(Qf, Kf, Vf, Of);
        f32_to_bf16<<<cvtGrid, blk, 0, stream>>>(Of, T0, nElem);
        gemm_bf16_wmma<<<gE, blk, 0, stream>>>(
            T0, t_wo + (size_t)l * E * E, t_bo + (size_t)l * E,
            Qf, nullptr, nullptr, M, E, E, 0);
        ln_residual<<<M, blk, 0, stream>>>(X, Qf, X,
            t_lng + (size_t)l * E, t_lnb + (size_t)l * E);

        // ===== FFN =====
        f32_to_bf16<<<cvtGrid, blk, 0, stream>>>(X, T0, nElem);
        gemm_bf16_wmma<<<gF, blk, 0, stream>>>(
            T0, f_w1 + (size_t)l * E * FF, f_b1 + (size_t)l * FF,
            nullptr, H1bf, nullptr, M, FF, E, GEMM_RELU | GEMM_OUTBF16);
        gemm_bf16_wmma<<<gE, blk, 0, stream>>>(
            H1bf, f_w2 + (size_t)l * FF * E, f_b2 + (size_t)l * E,
            Vf, nullptr, nullptr, M, E, FF, 0);
        ln_residual<<<M, blk, 0, stream>>>(X, Vf, X,
            f_lng + (size_t)l * E, f_lnb + (size_t)l * E);
    }

    // ---- head ----
    pool_mean<<<1024, blk, 0, stream>>>(X, pooled);
    float* out_f = (float*)d_out;
    head_kernel<<<16 * 63, blk, 0, stream>>>(pooled, h_lng, h_lnb, h_w, h_b, out_f);
    copy_f32<<<cvtGrid, blk, 0, stream>>>(X, out_f + 16 * 63 * 64, nElem);
}
